// AnomalyEncoder_86285892976937
// MI455X (gfx1250) — compile-verified, matrix operand order checked by reading, not run on hardware
//
#include <hip/hip_runtime.h>
#include <hip/hip_bf16.h>

#define BB    128
#define TT    2048
#define NE    8
#define NOUT  32
#define NHID  64
#define KC    8      // spline coefficients per input feature
#define NCHO  96     // OUT + HID output channels
#define CTILE 256

typedef __attribute__((ext_vector_type(16))) _Float16 v16h;
typedef __attribute__((ext_vector_type(8)))  float    v8f;

__device__ __forceinline__ float sigmoidf_(float x) { return 1.0f / (1.0f + __expf(-x)); }

// ---------------------------------------------------------------------------
// KAN-MoE: per-token (input feature dim == 1) dense softmax-gated expert mix.
// out[o] = sum_e gate[e] * ( silu(x)*base_w[e,o] + sum_k B_k(x)*spline_w[e,o,k] )
// ---------------------------------------------------------------------------
template <typename OutT>
__device__ __forceinline__ void kan_token(float x, const float* gw, const float* gb,
                                          const float* bw, const float* sw, OutT* out)
{
  // softmax gate over 8 experts
  float logits[NE]; float m = -1e30f;
#pragma unroll
  for (int e = 0; e < NE; e++) { logits[e] = x * gw[e] + gb[e]; m = fmaxf(m, logits[e]); }
  float gate[NE]; float s = 0.0f;
#pragma unroll
  for (int e = 0; e < NE; e++) { gate[e] = __expf(logits[e] - m); s += gate[e]; }
  float invs = 1.0f / s;
#pragma unroll
  for (int e = 0; e < NE; e++) gate[e] *= invs;

  // Cox-de Boor B-spline basis, ORDER=3, GRID=5, uniform knots on [-2.2, 2.2]
  float t[12];
#pragma unroll
  for (int j = 0; j < 12; j++) t[j] = 0.4f * (float)(j - 3) - 1.0f;
  float b[11];
#pragma unroll
  for (int j = 0; j < 11; j++) b[j] = (x >= t[j] && x < t[j + 1]) ? 1.0f : 0.0f;
#pragma unroll
  for (int k = 1; k <= 3; k++) {
    float invd = 1.0f / (0.4f * (float)k);
#pragma unroll
    for (int j = 0; j < 11 - k; j++)
      b[j] = (x - t[j]) * invd * b[j] + (t[j + k + 1] - x) * invd * b[j + 1];
  }

  float si = x * sigmoidf_(x);  // silu
#pragma unroll 4
  for (int o = 0; o < NOUT; o++) {
    float acc = 0.0f;
#pragma unroll
    for (int e = 0; e < NE; e++) {
      float v = si * bw[e * NOUT + o];
      const float* swp = &sw[(e * NOUT + o) * KC];
#pragma unroll
      for (int kk = 0; kk < KC; kk++) v += b[kk] * swp[kk];
      acc += gate[e] * v;
    }
    out[o] = (OutT)acc;
  }
}

__global__ __launch_bounds__(256) void kan_moe_kernel(
    const float* __restrict__ a, const float* __restrict__ dres,
    const float* __restrict__ gwa, const float* __restrict__ gba,
    const float* __restrict__ bwa, const float* __restrict__ swa,
    const float* __restrict__ gwd, const float* __restrict__ gbd,
    const float* __restrict__ bwd, const float* __restrict__ swd,
    float* __restrict__ fa, _Float16* __restrict__ fd16)
{
  __shared__ float s_gwa[NE], s_gba[NE], s_bwa[NE * NOUT], s_swa[NE * NOUT * KC];
  __shared__ float s_gwd[NE], s_gbd[NE], s_bwd[NE * NOUT], s_swd[NE * NOUT * KC];
  int tid = threadIdx.x;
  if (tid < NE) { s_gwa[tid] = gwa[tid]; s_gba[tid] = gba[tid];
                  s_gwd[tid] = gwd[tid]; s_gbd[tid] = gbd[tid]; }
  if (tid < NE * NOUT) { s_bwa[tid] = bwa[tid]; s_bwd[tid] = bwd[tid]; }
  for (int i = tid; i < NE * NOUT * KC; i += 256) { s_swa[i] = swa[i]; s_swd[i] = swd[i]; }
  __syncthreads();

  int idx = blockIdx.x * 256 + tid;  // token id in [0, B*T)
  float xa = a[idx];
  float xd = dres[idx];
  size_t base = (size_t)idx * NOUT;
  kan_token<float>(xa, s_gwa, s_gba, s_bwa, s_swa, fa + base);
  kan_token<_Float16>(xd, s_gwd, s_gbd, s_bwd, s_swd, fd16 + base);
}

// ---------------------------------------------------------------------------
// TemporalBlock: Conv1d(32->32, k=5, SAME) + ReLU, writes output channels 0..31
// ---------------------------------------------------------------------------
__global__ __launch_bounds__(256) void conv_relu_kernel(
    const float* __restrict__ fa, const float* __restrict__ cw,
    const float* __restrict__ cb, float* __restrict__ out)
{
  __shared__ float s_x[(CTILE + 4) * NOUT];
  __shared__ float s_w[5 * NOUT * NOUT];   // transposed to [k][ic][oc] (bank-friendly)
  __shared__ float s_b[NOUT];
  int b  = blockIdx.x >> 3;
  int t0 = (blockIdx.x & 7) * CTILE;
  int tid = threadIdx.x;

  for (int i = tid; i < (CTILE + 4) * NOUT; i += 256) {
    int tt = t0 + (i >> 5) - 2; int ic = i & 31;
    s_x[i] = (tt >= 0 && tt < TT) ? fa[((size_t)b * TT + tt) * NOUT + ic] : 0.0f;
  }
  for (int i = tid; i < 5 * NOUT * NOUT; i += 256) {
    int oc = i / 160; int rem = i - oc * 160; int ic = rem / 5; int k = rem - ic * 5;
    s_w[(k * NOUT + ic) * NOUT + oc] = cw[i];
  }
  if (tid < NOUT) s_b[tid] = cb[tid];
  __syncthreads();

  int oc = tid & 31, ts = tid >> 5;
  for (int tl = ts; tl < CTILE; tl += 8) {
    float acc = s_b[oc];
#pragma unroll
    for (int k = 0; k < 5; k++) {
      const float* xr = &s_x[(tl + k) * NOUT];
      const float* wr = &s_w[k * NOUT * NOUT];
#pragma unroll
      for (int ic = 0; ic < NOUT; ic++) acc = fmaf(xr[ic], wr[ic * NOUT + oc], acc);
    }
    out[((size_t)b * TT + (t0 + tl)) * NCHO + oc] = fmaxf(acc, 0.0f);
  }
}

// ---------------------------------------------------------------------------
// GRU: fused input projection + recurrence using v_wmma_f32_16x16x32_f16.
// One workgroup = 16 batch rows (M=16), 4 waves; each wave owns 3 N-tiles of
// the 192 gate outputs. Per step per wave: 3 WMMA (xp, K=32) + 6 WMMA (hp, K=64).
// Writes output channels 32..95.
// ---------------------------------------------------------------------------
// f16 A/B operand lane layout (ISA 7.12.2): element e of the v16h for a lane
// in half h maps to K index:
__device__ __forceinline__ int wk_of(int e, int half) {
  int v = e >> 1;
  return ((v & 4) ? 16 : 0) + half * 8 + ((v & 3) << 1) + (e & 1);
}

__global__ __launch_bounds__(128) void gru_wmma_kernel(
    const _Float16* __restrict__ fd16, const float* __restrict__ w_ih,
    const float* __restrict__ w_hh, const float* __restrict__ b_ih,
    const float* __restrict__ b_hh, float* __restrict__ out)
{
  __shared__ _Float16 s_x[16][32];     // x tile for current t (A operand feed)
  __shared__ _Float16 s_h16[16][64];   // h in f16 (A operand feed)
  __shared__ float    s_h[16][64];     // h master copy in f32
  __shared__ float    s_xp[16][192];   // x @ W_ih^T + b_ih
  __shared__ float    s_hp[16][192];   // h @ W_hh^T + b_hh

  int bg   = blockIdx.x;          // group of 16 batch rows
  int tid  = threadIdx.x;
  int lane = tid & 31, wv = tid >> 5;
  int half = lane >> 4, r16 = lane & 15;

  // preload this wave's weight tiles (B operands: B[k][n] = W[n_global][k],
  // loaded as W rows in the A-style layout with lane->n, elem->k)
  v16h wih[3], whh0[3], whh1[3];
  float bihv[3], bhhv[3];
#pragma unroll
  for (int j = 0; j < 3; j++) {
    int ng = (wv * 3 + j) * 16 + r16;
    bihv[j] = b_ih[ng];
    bhhv[j] = b_hh[ng];
#pragma unroll
    for (int e = 0; e < 16; e++) {
      int k = wk_of(e, half);
      wih[j][e]  = (_Float16)w_ih[ng * 32 + k];
      whh0[j][e] = (_Float16)w_hh[ng * 64 + k];
      whh1[j][e] = (_Float16)w_hh[ng * 64 + 32 + k];
    }
  }
  // h0 = 0
  for (int i = tid; i < 16 * 64; i += 128) {
    int r = i >> 6, c = i & 63;
    s_h[r][c] = 0.0f;
    s_h16[r][c] = (_Float16)0.0f;
  }
  __syncthreads();

  int rr = tid >> 3;              // 0..15 : row handled by this thread for x staging
  int kq = (tid & 7) * 4;         // 4 contiguous f16 per thread
  const _Float16* xg = fd16 + ((size_t)(bg * 16 + rr) * TT) * 32 + kq;

#pragma unroll 1
  for (int t = 0; t < TT; t++) {
    // stage x tile [16 x 32] f16 into LDS; prefetch next step
    const _Float16* xp_g = xg + (size_t)t * 32;
    s_x[rr][kq + 0] = xp_g[0]; s_x[rr][kq + 1] = xp_g[1];
    s_x[rr][kq + 2] = xp_g[2]; s_x[rr][kq + 3] = xp_g[3];
    __builtin_prefetch(xp_g + 32, 0, 1);
    __syncthreads();

    // gather A operands from LDS in the f16 16x32 layout
    v16h ax, ah0, ah1;
#pragma unroll
    for (int e = 0; e < 16; e++) {
      int k = wk_of(e, half);
      ax[e]  = s_x[r16][k];
      ah0[e] = s_h16[r16][k];
      ah1[e] = s_h16[r16][32 + k];
    }

#pragma unroll
    for (int j = 0; j < 3; j++) {
      v8f xc, hc;
#pragma unroll
      for (int i = 0; i < 8; i++) { xc[i] = bihv[j]; hc[i] = bhhv[j]; }
      xc = __builtin_amdgcn_wmma_f32_16x16x32_f16(false, ax,  false, wih[j],  (short)0, xc, false, false);
      hc = __builtin_amdgcn_wmma_f32_16x16x32_f16(false, ah0, false, whh0[j], (short)0, hc, false, false);
      hc = __builtin_amdgcn_wmma_f32_16x16x32_f16(false, ah1, false, whh1[j], (short)0, hc, false, false);
      int nb = (wv * 3 + j) * 16 + r16;
#pragma unroll
      for (int i = 0; i < 8; i++) {     // C/D layout: VGPR i -> M = i + 8*half, N = lane&15
        int M = i + 8 * half;
        s_xp[M][nb] = xc[i];
        s_hp[M][nb] = hc[i];
      }
    }
    __syncthreads();

    // gates: r,z,n in PyTorch order; 1024 (row,hid) items over 128 threads,
    // hid contiguous across lanes for conflict-free LDS + coalesced stores
#pragma unroll
    for (int s2 = 0; s2 < 8; s2++) {
      int idx = s2 * 128 + tid;
      int row = idx >> 6, hid = idx & 63;
      float rg = sigmoidf_(s_xp[row][hid]       + s_hp[row][hid]);
      float zg = sigmoidf_(s_xp[row][64 + hid]  + s_hp[row][64 + hid]);
      float nn = tanhf(s_xp[row][128 + hid] + rg * s_hp[row][128 + hid]);
      float hv = (1.0f - zg) * nn + zg * s_h[row][hid];
      s_h[row][hid] = hv;
      s_h16[row][hid] = (_Float16)hv;
      out[((size_t)(bg * 16 + row) * TT + t) * NCHO + 32 + hid] = hv;
    }
    __syncthreads();
  }
}

// ---------------------------------------------------------------------------
extern "C" void kernel_launch(void* const* d_in, const int* in_sizes, int n_in,
                              void* d_out, int out_size, void* d_ws, size_t ws_size,
                              hipStream_t stream)
{
  const float* a    = (const float*)d_in[0];
  const float* dres = (const float*)d_in[1];
  const float* gwa  = (const float*)d_in[2];
  const float* gba  = (const float*)d_in[3];
  const float* bwa  = (const float*)d_in[4];
  const float* swa  = (const float*)d_in[5];
  const float* gwd  = (const float*)d_in[6];
  const float* gbd  = (const float*)d_in[7];
  const float* bwd  = (const float*)d_in[8];
  const float* swd  = (const float*)d_in[9];
  const float* cw   = (const float*)d_in[10];
  const float* cb   = (const float*)d_in[11];
  const float* wih  = (const float*)d_in[12];
  const float* whh  = (const float*)d_in[13];
  const float* bih  = (const float*)d_in[14];
  const float* bhh  = (const float*)d_in[15];
  float* out = (float*)d_out;

  float*    fa   = (float*)d_ws;                                             // 32 MB
  _Float16* fd16 = (_Float16*)((char*)d_ws + (size_t)BB * TT * NOUT * 4);    // 16 MB

  kan_moe_kernel<<<(BB * TT) / 256, 256, 0, stream>>>(
      a, dres, gwa, gba, bwa, swa, gwd, gbd, bwd, swd, fa, fd16);
  conv_relu_kernel<<<BB * (TT / CTILE), 256, 0, stream>>>(fa, cw, cb, out);
  gru_wmma_kernel<<<BB / 16, 128, 0, stream>>>(fd16, wih, whh, bih, bhh, out);
}
